// Discriminator_10642928959686
// MI455X (gfx1250) — compile-verified, matrix operand order checked by reading, not run on hardware
//
#include <hip/hip_runtime.h>
#include <hip/hip_bf16.h>
#include <math.h>

typedef __attribute__((ext_vector_type(16))) _Float16 v16h;
typedef __attribute__((ext_vector_type(8)))  _Float16 v8h;
typedef __attribute__((ext_vector_type(8)))  float    v8f;
typedef __attribute__((ext_vector_type(4)))  float    v4f;

#define LEAKY(v) ((v) > 0.f ? (v) : 0.2f * (v))
#define CAT16(lo, hi) __builtin_shufflevector((lo), (hi), 0,1,2,3,4,5,6,7,8,9,10,11,12,13,14,15)

// ---------------------------------------------------------------------------
// Kernel 1: layout_bbox -> NCHW f16 (64,7,108,108)  (feeds conv1 A operand)
// ---------------------------------------------------------------------------
__global__ void layout_kernel(const float* __restrict__ img, _Float16* __restrict__ out) {
    const int HW = 108 * 108;
    int idx = blockIdx.x * blockDim.x + threadIdx.x;
    if (idx >= 64 * HW) return;
    int b = idx / HW;
    int r = idx % HW;
    int y = r / 108, x = r % 108;
    float fx = (float)x, fy = (float)y;
    float best[7] = {0.f, 0.f, 0.f, 0.f, 0.f, 0.f, 0.f};
    const float* base = img + (size_t)b * 16 * 11;
    for (int o = 0; o < 16; ++o) {
        const float* p = base + o * 11;
        float xc = p[0] * 108.f, yc = p[1] * 108.f;
        float w  = p[2] * 108.f, h  = p[3] * 108.f;
        float x1 = xc - 0.5f * w, x2 = xc + 0.5f * w;
        float y1 = yc - 0.5f * h, y2 = yc + 0.5f * h;
        float x1d = fx - x1, x2d = x2 - fx;
        float y1d = fy - y1, y2d = y2 - fy;
        float yb = fminf(fmaxf(y1d, 0.f), 1.f) * fminf(fmaxf(y2d, 0.f), 1.f);
        float xb = fminf(fmaxf(x1d, 0.f), 1.f) * fminf(fmaxf(x2d, 0.f), 1.f);
        float l1 = fmaxf(1.f - fabsf(x1d), 0.f) * yb;
        float l2 = fmaxf(1.f - fabsf(x2d), 0.f) * yb;
        float l3 = fmaxf(1.f - fabsf(y1d), 0.f) * xb;
        float l4 = fmaxf(1.f - fabsf(y2d), 0.f) * xb;
        float xy = fmaxf(fmaxf(l1, l2), fmaxf(l3, l4));
        #pragma unroll
        for (int c = 0; c < 7; ++c)
            best[c] = fmaxf(best[c], xy * p[4 + c]);
    }
    #pragma unroll
    for (int c = 0; c < 7; ++c)
        out[(((size_t)b * 7 + c) * 108 + y) * 108 + x] = (_Float16)best[c];
}

// ---------------------------------------------------------------------------
// Weight repack: OIHW f32 (N,Cin,5,5) -> f16 [n][ci*32 + r], r<25 valid, pad 0.
// One WMMA K-chunk == one input channel; B fragments become 2x16B loads.
// ---------------------------------------------------------------------------
__global__ void repack_w_kernel(const float* __restrict__ w, _Float16* __restrict__ wp,
                                int total) {
    int idx = blockIdx.x * blockDim.x + threadIdx.x;
    if (idx >= total) return;
    int r  = idx & 31;
    int nc = idx >> 5;                   // n*Cin + ci
    wp[idx] = (_Float16)((r < 25) ? w[nc * 25 + r] : 0.f);
}

// ---------------------------------------------------------------------------
// Kernel 2: conv1 implicit GEMM, WMMA. M=173056, N=32, Kpad=224 (7 chunks)
// Out raw f32 NCHW (64,32,52,52)
// ---------------------------------------------------------------------------
__global__ void conv1_kernel(const _Float16* __restrict__ in, const _Float16* __restrict__ wp,
                             const float* __restrict__ bias, float* __restrict__ out) {
    const int lane  = threadIdx.x & 31;
    const int wavei = threadIdx.x >> 5;
    const int mtile = blockIdx.x * 4 + wavei;
    const int row = lane & 15;
    const int col = lane & 15;
    const int kb  = (lane >> 4) << 3;            // 0 or 8
    const int m   = mtile * 16 + row;
    const int b   = m / 2704;
    const int rem = m % 2704;
    const int iy  = (rem / 52) * 2, ix = (rem % 52) * 2;
    const _Float16* inb = in + (size_t)b * 7 * 11664;

    // gather offsets: fixed per lane, hoisted out of the K loop.
    // invalid r (>=25) clamps to 0 -- the packed weight there is 0, so the
    // WMMA product contributes nothing.
    int offs[16];
    #pragma unroll
    for (int i = 0; i < 16; ++i) {
        int r = kb + i + ((i >> 3) << 3);
        offs[i] = (r < 25) ? ((iy + r / 5) * 108 + ix + r % 5) : 0;
    }
    const _Float16* wn0 = wp + (size_t)col * 224;
    const _Float16* wn1 = wp + (size_t)(col + 16) * 224;

    v8f acc0 = {}; v8f acc1 = {};
    for (int kc = 0; kc < 7; ++kc) {
        const _Float16* plane = inb + kc * 11664;
        v16h a;
        #pragma unroll
        for (int i = 0; i < 16; ++i) a[i] = plane[offs[i]];
        v16h b0 = CAT16(*(const v8h*)(wn0 + kc * 32 + kb), *(const v8h*)(wn0 + kc * 32 + kb + 16));
        v16h b1 = CAT16(*(const v8h*)(wn1 + kc * 32 + kb), *(const v8h*)(wn1 + kc * 32 + kb + 16));
        acc0 = __builtin_amdgcn_wmma_f32_16x16x32_f16(false, a, false, b0, (short)0, acc0, false, false);
        acc1 = __builtin_amdgcn_wmma_f32_16x16x32_f16(false, a, false, b1, (short)0, acc1, false, false);
    }
    const int rbase = (lane >> 4) << 3;
    #pragma unroll
    for (int v = 0; v < 8; ++v) {
        int mm = mtile * 16 + rbase + v;
        int bb = mm / 2704, rr = mm % 2704;
        int yy = rr / 52, xx = rr % 52;
        out[(((size_t)bb * 32 + col) * 52 + yy) * 52 + xx]      = acc0[v] + bias[col];
        out[(((size_t)bb * 32 + col + 16) * 52 + yy) * 52 + xx] = acc1[v] + bias[col + 16];
    }
}

// ---------------------------------------------------------------------------
// BN stats: x layout [b][c][S], one block (256 thr) per channel.
// ---------------------------------------------------------------------------
__global__ void bn_stats_kernel(const float* __restrict__ x, float* __restrict__ stats,
                                int C, int S, int Bn) {
    const int c = blockIdx.x;
    const int N = Bn * S;
    float sum = 0.f, sq = 0.f;
    for (int i = threadIdx.x; i < N; i += blockDim.x) {
        float v = x[((size_t)(i / S) * C + c) * S + (i % S)];
        sum += v; sq += v * v;
    }
    __shared__ float sh[512];
    sh[threadIdx.x] = sum; sh[threadIdx.x + 256] = sq;
    __syncthreads();
    for (int off = 128; off > 0; off >>= 1) {
        if (threadIdx.x < off) {
            sh[threadIdx.x]       += sh[threadIdx.x + off];
            sh[threadIdx.x + 256] += sh[threadIdx.x + 256 + off];
        }
        __syncthreads();
    }
    if (threadIdx.x == 0) {
        float m = sh[0] / (float)N;
        float var = sh[256] / (float)N - m * m;
        stats[c] = m;
        stats[C + c] = rsqrtf(var + 1e-5f);
    }
}

// ---------------------------------------------------------------------------
// BN apply + leaky-relu, emit f16 (next stage's WMMA A operand)
// ---------------------------------------------------------------------------
__global__ void bn_apply_kernel(const float* __restrict__ x, _Float16* __restrict__ y,
                                const float* __restrict__ stats, const float* __restrict__ g,
                                const float* __restrict__ beta, int C, int S, unsigned total) {
    unsigned i = blockIdx.x * blockDim.x + threadIdx.x;
    if (i >= total) return;
    int c = (int)((i / (unsigned)S) % (unsigned)C);
    float v = (x[i] - stats[c]) * stats[C + c] * g[c] + beta[c];
    y[i] = (_Float16)LEAKY(v);
}

// ---------------------------------------------------------------------------
// Kernel 5: conv2 implicit GEMM, WMMA. M=36864, N=64, Kpad=1024 (32 chunks)
// A: f16 NCHW act1 (64,32,52,52); out f32 [b][c*576+s] (fc1 flatten order)
// ---------------------------------------------------------------------------
__global__ void conv2_kernel(const _Float16* __restrict__ act1, const _Float16* __restrict__ wp,
                             const float* __restrict__ bias, float* __restrict__ out) {
    const int lane  = threadIdx.x & 31;
    const int wavei = threadIdx.x >> 5;
    const int mtile = blockIdx.x * 4 + wavei;
    const int row = lane & 15;
    const int col = lane & 15;
    const int kb  = (lane >> 4) << 3;
    const int m   = mtile * 16 + row;
    const int b   = m / 576;
    const int s   = m % 576;
    const int iy  = (s / 24) * 2, ix = (s % 24) * 2;
    const _Float16* inb = act1 + (size_t)b * 32 * 2704;

    int offs[16];
    #pragma unroll
    for (int i = 0; i < 16; ++i) {
        int r = kb + i + ((i >> 3) << 3);
        offs[i] = (r < 25) ? ((iy + r / 5) * 52 + ix + r % 5) : 0;
    }
    const _Float16* wn[4];
    #pragma unroll
    for (int t = 0; t < 4; ++t) wn[t] = wp + (size_t)(t * 16 + col) * 1024;

    v8f acc[4] = {{}, {}, {}, {}};
    for (int kc = 0; kc < 32; ++kc) {
        const _Float16* plane = inb + kc * 2704;
        v16h a;
        #pragma unroll
        for (int i = 0; i < 16; ++i) a[i] = plane[offs[i]];
        #pragma unroll
        for (int t = 0; t < 4; ++t) {
            v16h bf = CAT16(*(const v8h*)(wn[t] + kc * 32 + kb),
                            *(const v8h*)(wn[t] + kc * 32 + kb + 16));
            acc[t] = __builtin_amdgcn_wmma_f32_16x16x32_f16(false, a, false, bf, (short)0, acc[t], false, false);
        }
    }
    const int rbase = (lane >> 4) << 3;
    #pragma unroll
    for (int v = 0; v < 8; ++v) {
        int mm = mtile * 16 + rbase + v;
        int bb = mm / 576, ss = mm % 576;
        #pragma unroll
        for (int t = 0; t < 4; ++t) {
            int n = t * 16 + col;
            out[(size_t)bb * 36864 + n * 576 + ss] = acc[t][v] + bias[n];
        }
    }
}

// ---------------------------------------------------------------------------
// Kernel 8: fc1 WMMA GEMM, K-split 16. M=64, N=512, K=36864.
// Streams the 75.5 MB f32 weight matrix (the roofline term) once, converting
// to f16 in-register; 2048 waves spread the HBM read.
// ---------------------------------------------------------------------------
__global__ void fc1_kernel(const _Float16* __restrict__ act2, const float* __restrict__ w,
                           float* __restrict__ partial) {
    const int lane  = threadIdx.x & 31;
    const int wavei = threadIdx.x >> 5;
    const int wid   = blockIdx.x * 4 + wavei;   // 0..2047
    const int kslice = wid >> 7;                // 0..15
    const int rem    = wid & 127;
    const int ntile  = rem >> 2;                // 0..31
    const int mtile  = rem & 3;                 // 0..3
    const int row = lane & 15;
    const int col = lane & 15;
    const int kb  = (lane >> 4) << 3;
    const _Float16* arow = act2 + (size_t)(mtile * 16 + row) * 36864;
    const float*    wrow = w    + (size_t)(ntile * 16 + col) * 36864;
    const int kstart = kslice * 2304;

    v8f acc = {};
    for (int kc = 0; kc < 72; ++kc) {
        int k0 = kstart + kc * 32 + kb;
        v16h a = CAT16(*(const v8h*)(arow + k0), *(const v8h*)(arow + k0 + 16));
        v4f w0 = *(const v4f*)(wrow + k0);
        v4f w1 = *(const v4f*)(wrow + k0 + 4);
        v4f w2 = *(const v4f*)(wrow + k0 + 16);
        v4f w3 = *(const v4f*)(wrow + k0 + 20);
        v16h bf;
        #pragma unroll
        for (int i = 0; i < 4; ++i) {
            bf[i]      = (_Float16)w0[i];
            bf[i + 4]  = (_Float16)w1[i];
            bf[i + 8]  = (_Float16)w2[i];
            bf[i + 12] = (_Float16)w3[i];
        }
        acc = __builtin_amdgcn_wmma_f32_16x16x32_f16(false, a, false, bf, (short)0, acc, false, false);
    }
    const int rbase = (lane >> 4) << 3;
    float* p = partial + (((size_t)kslice * 4 + mtile) * 32 + ntile) * 256;
    #pragma unroll
    for (int v = 0; v < 8; ++v)
        p[(rbase + v) * 16 + col] = acc[v];
}

__global__ void fc1_reduce_kernel(const float* __restrict__ partial,
                                  const float* __restrict__ bias, float* __restrict__ out) {
    int idx = blockIdx.x * blockDim.x + threadIdx.x;   // 0..32767
    if (idx >= 64 * 512) return;
    int m = idx >> 9, n = idx & 511;
    int mtile = m >> 4, row = m & 15;
    int ntile = n >> 4, col = n & 15;
    float s = bias[n];
    for (int ks = 0; ks < 16; ++ks)
        s += partial[(((size_t)ks * 4 + mtile) * 32 + ntile) * 256 + row * 16 + col];
    out[idx] = s;
}

// ---------------------------------------------------------------------------
// Kernel 10: BN over batch (per feature) + leaky + fc2 + sigmoid. 1 block x512.
// ---------------------------------------------------------------------------
__global__ void head_kernel(const float* __restrict__ fc1out, const float* __restrict__ g,
                            const float* __restrict__ beta, const float* __restrict__ w2,
                            const float* __restrict__ b2, float* __restrict__ out) {
    const int j = threadIdx.x;   // 0..511
    float sum = 0.f, sq = 0.f;
    for (int b = 0; b < 64; ++b) {
        float v = fc1out[b * 512 + j];
        sum += v; sq += v * v;
    }
    float m  = sum * (1.f / 64.f);
    float rs = rsqrtf(sq * (1.f / 64.f) - m * m + 1e-5f);
    float gg = g[j] * rs, bb = beta[j], wj = w2[j];
    __shared__ float sh[512];
    for (int b = 0; b < 64; ++b) {
        float v = (fc1out[b * 512 + j] - m) * gg + bb;
        v = LEAKY(v);
        sh[j] = v * wj;
        __syncthreads();
        for (int off = 256; off > 0; off >>= 1) {
            if (j < off) sh[j] += sh[j + off];
            __syncthreads();
        }
        if (j == 0) out[b] = 1.f / (1.f + expf(-(sh[0] + b2[0])));
        __syncthreads();
    }
}

// ---------------------------------------------------------------------------
extern "C" void kernel_launch(void* const* d_in, const int* in_sizes, int n_in,
                              void* d_out, int out_size, void* d_ws, size_t ws_size,
                              hipStream_t stream) {
    const float* image   = (const float*)d_in[0];
    const float* conv1_w = (const float*)d_in[1];
    const float* conv1_b = (const float*)d_in[2];
    const float* bn1_g   = (const float*)d_in[3];
    const float* bn1_b   = (const float*)d_in[4];
    const float* conv2_w = (const float*)d_in[5];
    const float* conv2_b = (const float*)d_in[6];
    const float* bn2_g   = (const float*)d_in[7];
    const float* bn2_b   = (const float*)d_in[8];
    const float* fc1_w   = (const float*)d_in[9];
    const float* fc1_b   = (const float*)d_in[10];
    const float* bn3_g   = (const float*)d_in[11];
    const float* bn3_b   = (const float*)d_in[12];
    const float* fc2_w   = (const float*)d_in[13];
    const float* fc2_b   = (const float*)d_in[14];
    float* out = (float*)d_out;

    char* ws = (char*)d_ws;
    size_t off = 0;
    auto alloc = [&](size_t bytes) {
        void* p = ws + off;
        off = (off + bytes + 255) & ~(size_t)255;
        return p;
    };
    _Float16* layout_h = (_Float16*)alloc((size_t)64 * 7 * 11664 * 2);    // 10.5 MB
    _Float16* wpack1   = (_Float16*)alloc((size_t)32 * 224 * 2);
    _Float16* wpack2   = (_Float16*)alloc((size_t)64 * 1024 * 2);
    float*    c1out    = (float*)   alloc((size_t)64 * 32 * 2704 * 4);    // 22.2 MB
    _Float16* act1     = (_Float16*)alloc((size_t)64 * 32 * 2704 * 2);    // 11.1 MB
    float*    c2out    = (float*)   alloc((size_t)64 * 36864 * 4);        //  9.4 MB
    _Float16* act2     = (_Float16*)alloc((size_t)64 * 36864 * 2);        //  4.7 MB
    float*    stats1   = (float*)   alloc(64 * 4);
    float*    stats2   = (float*)   alloc(128 * 4);
    float*    fc1part  = (float*)   alloc((size_t)16 * 4 * 32 * 256 * 4); //  2.1 MB
    float*    fc1out   = (float*)   alloc((size_t)64 * 512 * 4);

    // 0) repack conv weights into padded f16 fragment-friendly layout
    repack_w_kernel<<<(32 * 224 + 255) / 256, 256, 0, stream>>>(conv1_w, wpack1, 32 * 224);
    repack_w_kernel<<<(64 * 1024 + 255) / 256, 256, 0, stream>>>(conv2_w, wpack2, 64 * 1024);
    // 1) layout_bbox (f16 out)
    layout_kernel<<<(64 * 11664 + 255) / 256, 256, 0, stream>>>(image, layout_h);
    // 2) conv1 (WMMA): 10816 M-tiles, 4 waves/block
    conv1_kernel<<<2704, 128, 0, stream>>>(layout_h, wpack1, conv1_b, c1out);
    // 3) BN1 stats + apply
    bn_stats_kernel<<<32, 256, 0, stream>>>(c1out, stats1, 32, 2704, 64);
    {
        unsigned total = 64u * 32u * 2704u;
        bn_apply_kernel<<<(total + 255) / 256, 256, 0, stream>>>(
            c1out, act1, stats1, bn1_g, bn1_b, 32, 2704, total);
    }
    // 4) conv2 (WMMA): 2304 M-tiles
    conv2_kernel<<<576, 128, 0, stream>>>(act1, wpack2, conv2_b, c2out);
    // 5) BN2 stats + apply
    bn_stats_kernel<<<64, 256, 0, stream>>>(c2out, stats2, 64, 576, 64);
    {
        unsigned total = 64u * 36864u;
        bn_apply_kernel<<<(total + 255) / 256, 256, 0, stream>>>(
            c2out, act2, stats2, bn2_g, bn2_b, 64, 576, total);
    }
    // 6) fc1 (WMMA, 16-way K split) + deterministic reduction
    fc1_kernel<<<512, 128, 0, stream>>>(act2, fc1_w, fc1part);
    fc1_reduce_kernel<<<(64 * 512 + 255) / 256, 256, 0, stream>>>(fc1part, fc1_b, fc1out);
    // 7) BN3 + leaky + fc2 + sigmoid
    head_kernel<<<1, 512, 0, stream>>>(fc1out, bn3_g, bn3_b, fc2_w, fc2_b, out);
}
